// GlobalLinearAttention_46153718562973
// MI455X (gfx1250) — matrix operand order repaired and checked by reading, not compiled, not run on hardware
//
#include <hip/hip_runtime.h>
#include <cmath>

// Problem constants (from reference): B=4, N=16384, C=256, H=4, D=64
#define BATCH 4
#define SEQ   16384
#define CH    256
#define NHEAD 4
#define HDIM  64

typedef __bf16 bf16;
typedef __attribute__((ext_vector_type(16))) __bf16 v16bf;
typedef __attribute__((ext_vector_type(8)))  __bf16 bf16x8;
typedef __attribute__((ext_vector_type(8)))  float  v8f;
typedef unsigned int u32;
typedef __attribute__((ext_vector_type(2))) unsigned int u32x2;
typedef __attribute__((ext_vector_type(8))) unsigned int u32x8;

// ---------------------------------------------------------------------------
// WMMA: D = A(16x32 bf16) * B(32x16 bf16) + C(16x16 f32)
// ---------------------------------------------------------------------------
__device__ __forceinline__ v8f wmma_bf16(v16bf a, v16bf b, v8f c) {
  return __builtin_amdgcn_wmma_f32_16x16x32_bf16(
      false, a, false, b, (short)0, c, false, false);
}

// Pack two fp32 into one dword of two bf16 (lo, hi)
__device__ __forceinline__ u32 pack2bf(float lo, float hi) {
  const unsigned short a = __builtin_bit_cast(unsigned short, (bf16)lo);
  const unsigned short c = __builtin_bit_cast(unsigned short, (bf16)hi);
  return ((u32)c << 16) | a;
}

// Fragment-packed LDS layouts: each lane's 16 operand elements are one
// contiguous 32-byte run -> fragment load = 2x ds_load_b128, zero shuffles.
//
// A (16x32 MxK per ISA): lane = (m&15) + 16*halfA, halfA=(rl>>3)&1,
//                        elem = (rl&7) + 8*(rl>>4), rl = k&31
__device__ __forceinline__ int aidx(int m, int k) {
  return (((k >> 5) * 4 + (m >> 4)) * 32 + (m & 15) + 16 * ((k >> 3) & 1)) * 16
         + (k & 7) + 8 * ((k >> 4) & 1);
}
// B (32x16 KxN per ISA): lane = (n&15) + 16*((k>>3)&1), elem = (k&7) + 8*((k>>4)&1)
__device__ __forceinline__ int bidx(int k, int n) {
  return (((k >> 5) * 4 + (n >> 4)) * 32 + (n & 15) + 16 * ((k >> 3) & 1)) * 16
         + (k & 7) + 8 * ((k >> 4) & 1);
}

__device__ __forceinline__ v16bf load_frag_packed(const bf16* s, int tile) {
  const int lane = threadIdx.x & 31;
  const u32x8 t = *(const u32x8*)(s + (tile * 32 + lane) * 16);
  return __builtin_bit_cast(v16bf, t);
}

__device__ __forceinline__ float gelu_exact(float x) {
  return 0.5f * x * (1.0f + erff(x * 0.70710678118654752f));
}

// ===========================================================================
// Kernel 1: K/V projection + mask/gelu epilogue + KV = K^T V and Ksum.
// Grid: (SEQ/256, BATCH), 256 threads (8 waves); 4 subtiles of 64 rows.
// KV fragments accumulate in registers for the whole block; fp32 atomics out.
// ===========================================================================
__global__ __launch_bounds__(256)
void gla_kv_reduce(const float* __restrict__ x, const float* __restrict__ mask_y,
                   const float* __restrict__ Wk, const float* __restrict__ bk,
                   const float* __restrict__ Wv, const float* __restrict__ bv,
                   float* __restrict__ KV, float* __restrict__ Ksum) {
  __shared__ bf16  sX[8 * 4 * 32 * 16];    // 32 KB, A-packed x (64x256)
  __shared__ bf16  sWkB[4 * 32 * 16];      //  4 KB, B-packed Wk chunk (32x64)
  __shared__ bf16  sWvB[4 * 32 * 16];      //  4 KB, B-packed Wv chunk
  __shared__ bf16  sKT[2 * 4 * 32 * 16];   //  8 KB, A-packed K^T (64x64)
  __shared__ bf16  sVB[2 * 4 * 32 * 16];   //  8 KB, B-packed V   (64x64)
  __shared__ float sKsum[NHEAD * HDIM];    //  1 KB
  __shared__ float sMy[64];

  const int tid  = threadIdx.x;
  const int wave = tid >> 5, lane = tid & 31;
  const int b    = blockIdx.y;
  const int row_block = blockIdx.x * 256;
  const int fi  = wave >> 1;         // row/d tile (0..3)
  const int fj0 = (wave & 1) * 2;    // col/e tile base (0 or 2)
  const int cN  = lane & 15;
  const int hl  = lane >> 4;

  sKsum[tid] = 0.0f;                 // 256 == NHEAD*HDIM entries
  v8f kvacc[8] = {};                 // [head][frag]

  for (int sub = 0; sub < 4; ++sub) {
    const int row0 = row_block + sub * 64;

    // ---- stage x subtile into A-packed LDS (fp32 -> packed bf16) --------
    for (int i = tid; i < 64 * 64; i += 256) {            // 64 float4 / row
      const int r = i >> 6, c4 = (i & 63) << 2;
      const float* gp = x + ((size_t)b * SEQ + row0 + r) * CH + c4;
      const float4 v = *(const float4*)gp;
      if (sub < 3) __builtin_prefetch(gp + 64 * CH, 0, 0);
      u32x2 p;
      p.x = pack2bf(v.x, v.y);
      p.y = pack2bf(v.z, v.w);
      *(u32x2*)(&sX[aidx(r, c4)]) = p;
    }
    if (tid < 64) sMy[tid] = mask_y[(size_t)b * SEQ + row0 + tid];
    __syncthreads();

#pragma unroll
    for (int h = 0; h < NHEAD; ++h) {
      // ---- K/V projection (64x64 head slice) --------------------------
      v8f ak0 = {}, ak1 = {}, av0 = {}, av1 = {};
      for (int kk = 0; kk < CH; kk += 32) {
        for (int i = tid; i < 16 * 16; i += 256) {  // 16 k-pairs x 16 quads
          const int rp = i >> 4, c4 = (i & 15) << 2;
          const int k  = rp * 2;
          const float* w0 = Wk + (size_t)(kk + k) * CH + h * HDIM + c4;
          const float* w1 = w0 + CH;
          const float* v0 = Wv + (size_t)(kk + k) * CH + h * HDIM + c4;
          const float* v1 = v0 + CH;
#pragma unroll
          for (int t = 0; t < 4; ++t) {
            *(u32*)(&sWkB[bidx(k, c4 + t)]) = pack2bf(w0[t], w1[t]);
            *(u32*)(&sWvB[bidx(k, c4 + t)]) = pack2bf(v0[t], v1[t]);
          }
        }
        __syncthreads();
        const v16bf a  = load_frag_packed(sX, (kk >> 5) * 4 + fi);
        const v16bf b0 = load_frag_packed(sWkB, fj0);
        const v16bf b1 = load_frag_packed(sWkB, fj0 + 1);
        const v16bf c0 = load_frag_packed(sWvB, fj0);
        const v16bf c1 = load_frag_packed(sWvB, fj0 + 1);
        ak0 = wmma_bf16(a, b0, ak0);
        ak1 = wmma_bf16(a, b1, ak1);
        av0 = wmma_bf16(a, c0, av0);
        av1 = wmma_bf16(a, c1, av1);
        __syncthreads();
      }

      // ---- epilogue: K=gelu((p+bk)*my)+1 -> A-packed K^T;
      //                V=(p+bv)*my        -> B-packed V; Ksum via ds_add --
      {
        const float bk0v = bk[h * HDIM + fj0 * 16 + cN];
        const float bk1v = bk[h * HDIM + fj0 * 16 + cN + 16];
        const float bv0v = bv[h * HDIM + fj0 * 16 + cN];
        const float bv1v = bv[h * HDIM + fj0 * 16 + cN + 16];
        const int kc = fi >> 1;
        // K^T: m = d, k = row  ->  lane = cN+16*hl, e = r + 8*(fi&1)
        bf16* kt0 = &sKT[((kc * 4 + fj0    ) * 32 + cN + 16 * hl) * 16 + 8 * (fi & 1)];
        bf16* kt1 = &sKT[((kc * 4 + fj0 + 1) * 32 + cN + 16 * hl) * 16 + 8 * (fi & 1)];
        // V: k = row, n = e-col -> lane = cN+16*hl, e = r + 8*(fi&1)
        bf16* vb0 = &sVB[((kc * 4 + fj0    ) * 32 + cN + 16 * hl) * 16 + 8 * (fi & 1)];
        bf16* vb1 = &sVB[((kc * 4 + fj0 + 1) * 32 + cN + 16 * hl) * 16 + 8 * (fi & 1)];
        float ks0 = 0.0f, ks1 = 0.0f;
#pragma unroll
        for (int r = 0; r < 8; r += 2) {
          const int rowt = fi * 16 + r + 8 * hl;
          const float my0 = sMy[rowt], my1 = sMy[rowt + 1];
          const float k00 = gelu_exact((ak0[r]     + bk0v) * my0) + 1.0f;
          const float k01 = gelu_exact((ak0[r + 1] + bk0v) * my1) + 1.0f;
          const float k10 = gelu_exact((ak1[r]     + bk1v) * my0) + 1.0f;
          const float k11 = gelu_exact((ak1[r + 1] + bk1v) * my1) + 1.0f;
          const float v00 = (av0[r] + bv0v) * my0, v01 = (av0[r + 1] + bv0v) * my1;
          const float v10 = (av1[r] + bv1v) * my0, v11 = (av1[r + 1] + bv1v) * my1;
          *(u32*)(kt0 + r) = pack2bf(k00, k01);
          *(u32*)(kt1 + r) = pack2bf(k10, k11);
          *(u32*)(vb0 + r) = pack2bf(v00, v01);
          *(u32*)(vb1 + r) = pack2bf(v10, v11);
          ks0 += k00 + k01;
          ks1 += k10 + k11;
        }
        atomicAdd(&sKsum[h * HDIM + fj0 * 16 + cN],      ks0);
        atomicAdd(&sKsum[h * HDIM + fj0 * 16 + cN + 16], ks1);
      }
      __syncthreads();

      // ---- KV += K^T (64 x 64rows) * V (64rows x 64) ------------------
      {
        const v16bf at0 = load_frag_packed(sKT, fi);
        const v16bf at1 = load_frag_packed(sKT, 4 + fi);
        const v16bf b0a = load_frag_packed(sVB, fj0);
        const v16bf b0b = load_frag_packed(sVB, 4 + fj0);
        const v16bf b1a = load_frag_packed(sVB, fj0 + 1);
        const v16bf b1b = load_frag_packed(sVB, 4 + fj0 + 1);
        kvacc[h * 2]     = wmma_bf16(at0, b0a, kvacc[h * 2]);
        kvacc[h * 2]     = wmma_bf16(at1, b0b, kvacc[h * 2]);
        kvacc[h * 2 + 1] = wmma_bf16(at0, b1a, kvacc[h * 2 + 1]);
        kvacc[h * 2 + 1] = wmma_bf16(at1, b1b, kvacc[h * 2 + 1]);
      }
      __syncthreads();
    }
  }

  // ---- fp32 atomic accumulation into workspace --------------------------
#pragma unroll
  for (int h = 0; h < NHEAD; ++h) {
    float* kvbase = KV + (size_t)(b * NHEAD + h) * HDIM * HDIM;
    const int dbase = fi * 16 + 8 * hl;
    const int e0    = fj0 * 16 + cN;
#pragma unroll
    for (int r = 0; r < 8; ++r) {
      atomicAdd(kvbase + (dbase + r) * HDIM + e0,      kvacc[h * 2][r]);
      atomicAdd(kvbase + (dbase + r) * HDIM + e0 + 16, kvacc[h * 2 + 1][r]);
    }
  }
  atomicAdd(Ksum + (size_t)b * NHEAD * HDIM + tid, sKsum[tid]);
}

// ===========================================================================
// Kernel 2: Q projection + denom + attn = (Q KV)/denom * mx + out projection.
// Grid: (SEQ/64, BATCH), 256 threads. One 32KB A-packed region holds
// x -> Q -> attn in sequence (in-place overwrite at phase boundaries).
// ===========================================================================
__global__ __launch_bounds__(256)
void gla_attn_out(const float* __restrict__ x, const float* __restrict__ mask_x,
                  const float* __restrict__ Wq, const float* __restrict__ bq,
                  const float* __restrict__ Wp, const float* __restrict__ bp,
                  const float* __restrict__ KV, const float* __restrict__ Ksum,
                  float* __restrict__ out) {
  __shared__ bf16  sXQ[8 * 4 * 32 * 16];   // 32 KB, A-packed
  __shared__ bf16  sWB[16 * 32 * 16];      // 16 KB, B-packed chunk / KV
  __shared__ float sDenom[NHEAD * 64];
  __shared__ float sMx[64];

  const int tid  = threadIdx.x;
  const int wave = tid >> 5, lane = tid & 31;
  const int b    = blockIdx.y;
  const int row0 = blockIdx.x * 64;
  const int fi   = wave >> 1;
  const int fj0  = (wave & 1) * 2;
  const int cN   = lane & 15;
  const int hl   = lane >> 4;

  // ---- stage x tile (A-packed) -------------------------------------------
  for (int i = tid; i < 64 * 64; i += 256) {
    const int r = i >> 6, c4 = (i & 63) << 2;
    const float4 v = *(const float4*)(x + ((size_t)b * SEQ + row0 + r) * CH + c4);
    u32x2 p;
    p.x = pack2bf(v.x, v.y);
    p.y = pack2bf(v.z, v.w);
    *(u32x2*)(&sXQ[aidx(r, c4)]) = p;
  }
  if (tid < 64) sMx[tid] = mask_x[(size_t)b * SEQ + row0 + tid];
  __syncthreads();

  // ---- Q projection: wave owns col tiles {2w, 2w+1} ----------------------
  v8f qa[8] = {};
  for (int kk = 0; kk < CH; kk += 32) {
    for (int i = tid; i < 16 * 64; i += 256) {     // 16 k-pairs x 64 quads
      const int rp = i >> 6, c4 = (i & 63) << 2;
      const int k  = rp * 2;
      const float* w0 = Wq + (size_t)(kk + k) * CH + c4;
      const float* w1 = w0 + CH;
#pragma unroll
      for (int t = 0; t < 4; ++t)
        *(u32*)(&sWB[bidx(k, c4 + t)]) = pack2bf(w0[t], w1[t]);
    }
    __syncthreads();
    const v16bf b0 = load_frag_packed(sWB, 2 * wave);
    const v16bf b1 = load_frag_packed(sWB, 2 * wave + 1);
#pragma unroll
    for (int i2 = 0; i2 < 4; ++i2) {
      const v16bf a = load_frag_packed(sXQ, (kk >> 5) * 4 + i2);
      qa[i2 * 2]     = wmma_bf16(a, b0, qa[i2 * 2]);
      qa[i2 * 2 + 1] = wmma_bf16(a, b1, qa[i2 * 2 + 1]);
    }
    __syncthreads();
  }

  // ---- Q epilogue: gelu(p+bq)+1, overwrite x region (A-packed) -----------
#pragma unroll
  for (int jj = 0; jj < 2; ++jj) {
    const int jt  = 2 * wave + jj;
    const float bqv = bq[jt * 16 + cN];
    const int kcq   = jt >> 1;
    const int laneA = 16 * ((cN >> 3) & 1);
    const int e     = (cN & 7) + 8 * (jt & 1);
#pragma unroll
    for (int i2 = 0; i2 < 4; ++i2) {
#pragma unroll
      for (int r = 0; r < 8; ++r) {
        sXQ[((kcq * 4 + i2) * 32 + r + 8 * hl + laneA) * 16 + e] =
            (bf16)(gelu_exact(qa[i2 * 2 + jj][r] + bqv) + 1.0f);
      }
    }
  }
  __syncthreads();

  // ---- denom[row][h] = sum_d Q[row, h*64+d] * Ksum[b,h,d] ----------------
  {
    const int row = tid & 63, h = tid >> 6;
    const float* ks = Ksum + (size_t)(b * NHEAD + h) * HDIM;
    const int mi = row >> 4, m15 = row & 15;
    float s = 0.0f;
#pragma unroll
    for (int j8 = 0; j8 < 8; ++j8) {
      const int kc    = 2 * h + (j8 >> 2);
      const int lane8 = m15 + 16 * (j8 & 1);
      const int e0    = 8 * ((j8 >> 1) & 1);
      const bf16x8 qv = *(const bf16x8*)(&sXQ[((kc * 4 + mi) * 32 + lane8) * 16 + e0]);
#pragma unroll
      for (int t = 0; t < 8; ++t) s += (float)qv[t] * ks[j8 * 8 + t];
    }
    sDenom[h * 64 + row] = s;
  }
  __syncthreads();

  // ---- attn per head: (Q_h @ KV_h)/denom * mx, in place over Q_h ---------
  for (int h = 0; h < NHEAD; ++h) {
    const float* kvg = KV + (size_t)(b * NHEAD + h) * HDIM * HDIM;
    for (int i = tid; i < 32 * 16; i += 256) {       // 32 k-pairs x 16 quads
      const int rp = i >> 4, c4 = (i & 15) << 2;
      const int k  = rp * 2;
      const float4 r0 = *(const float4*)(kvg + (size_t)k * HDIM + c4);
      const float4 r1 = *(const float4*)(kvg + (size_t)(k + 1) * HDIM + c4);
      const float* p0 = (const float*)&r0;
      const float* p1 = (const float*)&r1;
#pragma unroll
      for (int t = 0; t < 4; ++t)
        *(u32*)(&sWB[bidx(k, c4 + t)]) = pack2bf(p0[t], p1[t]);
    }
    __syncthreads();

    v8f aa0 = {}, aa1 = {};
#pragma unroll
    for (int kc = 0; kc < 2; ++kc) {
      const v16bf a  = load_frag_packed(sXQ, (2 * h + kc) * 4 + fi);
      const v16bf b0 = load_frag_packed(sWB, kc * 4 + fj0);
      const v16bf b1 = load_frag_packed(sWB, kc * 4 + fj0 + 1);
      aa0 = wmma_bf16(a, b0, aa0);
      aa1 = wmma_bf16(a, b1, aa1);
    }
    __syncthreads();

    {
      const int kc2   = 2 * h + (fj0 >> 1);
      const int laneA = 16 * ((cN >> 3) & 1);
      const int e0    = cN & 7;
#pragma unroll
      for (int r = 0; r < 8; ++r) {
        const int rowt  = fi * 16 + r + 8 * hl;
        const float sc  = sMx[rowt] / sDenom[h * 64 + rowt];
        const int base  = ((kc2 * 4 + fi) * 32 + r + 8 * hl + laneA) * 16;
        sXQ[base + e0]     = (bf16)(aa0[r] * sc);
        sXQ[base + e0 + 8] = (bf16)(aa1[r] * sc);
      }
    }
    __syncthreads();
  }

  // ---- output projection: out = attn @ Wp + bp ---------------------------
  v8f oa[8] = {};
  for (int kk = 0; kk < CH; kk += 32) {
    for (int i = tid; i < 16 * 64; i += 256) {
      const int rp = i >> 6, c4 = (i & 63) << 2;
      const int k  = rp * 2;
      const float* w0 = Wp + (size_t)(kk + k) * CH + c4;
      const float* w1 = w0 + CH;
#pragma unroll
      for (int t = 0; t < 4; ++t)
        *(u32*)(&sWB[bidx(k, c4 + t)]) = pack2bf(w0[t], w1[t]);
    }
    __syncthreads();
    const v16bf b0 = load_frag_packed(sWB, 2 * wave);
    const v16bf b1 = load_frag_packed(sWB, 2 * wave + 1);
#pragma unroll
    for (int i2 = 0; i2 < 4; ++i2) {
      const v16bf a = load_frag_packed(sXQ, (kk >> 5) * 4 + i2);
      oa[i2 * 2]     = wmma_bf16(a, b0, oa[i2 * 2]);
      oa[i2 * 2 + 1] = wmma_bf16(a, b1, oa[i2 * 2 + 1]);
    }
    __syncthreads();
  }

  {
    const float bp0 = bp[(2 * wave) * 16 + cN];
    const float bp1 = bp[(2 * wave + 1) * 16 + cN];
#pragma unroll
    for (int i2 = 0; i2 < 4; ++i2) {
#pragma unroll
      for (int r = 0; r < 8; ++r) {
        const int rowt = i2 * 16 + r + 8 * hl;
        float* o = out + ((size_t)b * SEQ + row0 + rowt) * CH;
        o[(2 * wave) * 16 + cN]     = oa[i2 * 2][r]     + bp0;
        o[(2 * wave + 1) * 16 + cN] = oa[i2 * 2 + 1][r] + bp1;
      }
    }
  }
}

// ===========================================================================
extern "C" void kernel_launch(void* const* d_in, const int* in_sizes, int n_in,
                              void* d_out, int out_size, void* d_ws, size_t ws_size,
                              hipStream_t stream) {
  const float* x      = (const float*)d_in[0];
  const float* mask_x = (const float*)d_in[1];
  const float* mask_y = (const float*)d_in[2];
  const float* Wq     = (const float*)d_in[3];
  const float* bq     = (const float*)d_in[4];
  const float* Wk     = (const float*)d_in[5];
  const float* bk     = (const float*)d_in[6];
  const float* Wv     = (const float*)d_in[7];
  const float* bv     = (const float*)d_in[8];
  const float* Wp     = (const float*)d_in[9];
  const float* bp     = (const float*)d_in[10];
  float* out = (float*)d_out;

  float* KV   = (float*)d_ws;                              // B*H*64*64 fp32
  float* Ksum = KV + (size_t)BATCH * NHEAD * HDIM * HDIM;  // B*H*64 fp32
  const size_t acc_bytes =
      ((size_t)BATCH * NHEAD * HDIM * HDIM + (size_t)BATCH * NHEAD * HDIM) * sizeof(float);
  hipMemsetAsync(d_ws, 0, acc_bytes, stream);

  gla_kv_reduce<<<dim3(SEQ / 256, BATCH), 256, 0, stream>>>(
      x, mask_y, Wk, bk, Wv, bv, KV, Ksum);
  gla_attn_out<<<dim3(SEQ / 64, BATCH), 256, 0, stream>>>(
      x, mask_x, Wq, bq, Wp, bp, KV, Ksum, out);
}
